// ISTFT_34076270527008
// MI455X (gfx1250) — matrix operand order repaired and checked by reading, not compiled, not run on hardware
//
#include <hip/hip_runtime.h>
#include <hip/hip_bf16.h>
#include <math.h>

// ---------------- problem constants ----------------
#define BATCH     16
#define NBINS     513
#define TFRAMES   2048
#define NFFT      1024
#define HOP       256
#define WINLEN    1024
#define PAD       384                  // (WINLEN - HOP) / 2
#define OUT_T     524288               // (TFRAMES-1)*HOP + WINLEN - 2*PAD
#define NCOLS     (BATCH * TFRAMES)    // 32768 GEMM columns
#define KTOT      1026                 // 513 cos rows + 513 (-sin) rows
#define KTILES    33                   // ceil(1026/32) -> Kpad = 1056
#define MTILES    64                   // 1024 / 16
#define CTILES    2048                 // 32768 / 16
#define FRAG_ELTS 512                  // one 16x32 f16 fragment (32 lanes x 16 halves)

typedef __attribute__((ext_vector_type(16))) _Float16 v16h;
typedef __attribute__((ext_vector_type(8)))  float    v8f;

// ---------------- fragment K-index swizzles (per CDNA5 ISA 7.12.2) ----------------
// A (16-bit, 16x32): lanes 0-15: j0..7 -> K=j, j8..15 -> K=16+(j-8)
//                    lanes16-31: j0..7 -> K=8+j, j8..15 -> K=24+(j-8)
__device__ __forceinline__ int kmapA(int laneHi, int j) {
    if (!laneHi) return (j < 8) ? j : 16 + (j - 8);
    else         return (j < 8) ? 8 + j : 24 + (j - 8);
}
// B (16-bit, 32x16): same K order as A: lanes 0-15: j0..7 -> K=j, j8..15 -> K=16+(j-8); lanes 16-31: +8
__device__ __forceinline__ int kmapB(int laneHi, int j) {
    return kmapA(laneHi, j);
}

// ---------------- kernel 1: build swizzled f16 basis A[mtile][ktile][frag] ----------------
// A[n][k] = c_k * cos(2*pi*k*n/1024)          for k < 513
//         = -c_k' * sin(2*pi*k'*n/1024)       for 513 <= k < 1026 (k' = k-513)
//         = 0                                  for K padding
__global__ __launch_bounds__(256) void istft_build_basis(_Float16* __restrict__ Afrag) {
    const size_t idx = (size_t)blockIdx.x * 256 + threadIdx.x;   // 64*33*512 total
    const int e     = (int)(idx & 511);
    const int lane  = e >> 4;
    const int j     = e & 15;
    const int kt    = (int)((idx >> 9) % KTILES);
    const int mtile = (int)(idx / ((size_t)KTILES * FRAG_ELTS));
    const int n     = mtile * 16 + (lane & 15);
    const int k     = kt * 32 + kmapA(lane >> 4, j);

    const float step = 6.2831853071795864769f / 1024.0f;
    float v = 0.0f;
    if (k < NBINS) {
        const float c = (k == 0 || k == 512) ? 1.0f : 2.0f;
        const int m = (k * n) & 1023;                // exact phase reduction
        v = c * __cosf((float)m * step);
    } else if (k < KTOT) {
        const int kk = k - NBINS;
        const float c = (kk == 0 || kk == 512) ? 1.0f : 2.0f;
        const int m = (kk * n) & 1023;
        v = -c * __sinf((float)m * step);
    }
    Afrag[idx] = (_Float16)v;
}

// ---------------- kernel 2: swizzle spectra into f16 B fragments B[ctile][ktile][frag] ----------------
__global__ __launch_bounds__(256) void istft_convert_spec(const float* __restrict__ re,
                                                          const float* __restrict__ im,
                                                          _Float16* __restrict__ Bfrag) {
    const size_t idx = (size_t)blockIdx.x * 256 + threadIdx.x;   // 2048*33*512 total
    const int e     = (int)(idx & 511);
    const int lane  = e >> 4;
    const int j     = e & 15;
    const int kt    = (int)((idx >> 9) % KTILES);
    const int ctile = (int)(idx / ((size_t)KTILES * FRAG_ELTS));
    const int col   = ctile * 16 + (lane & 15);
    const int k     = kt * 32 + kmapB(lane >> 4, j);
    const int b     = col >> 11;          // / TFRAMES
    const int t     = col & 2047;         // % TFRAMES

    float v = 0.0f;
    if (k < NBINS) {
        v = re[((size_t)b * NBINS + k) * TFRAMES + t];
    } else if (k < KTOT) {
        v = im[((size_t)b * NBINS + (k - NBINS)) * TFRAMES + t];
    }
    Bfrag[idx] = (_Float16)v;
}

// ---------------- kernel 3: WMMA GEMM -> windowed frames[b][t][n] ----------------
// Workgroup tile: 128 rows (M) x 64 cols (N). 8 waves, each wave = 2x2 16x16 tiles.
// Result scaled by window[n]/1024 and transposed via LDS so frames are row-contiguous
// in n for coalesced stores and coalesced overlap-add reads.
__global__ __launch_bounds__(256) void istft_gemm_wmma(const _Float16* __restrict__ Afrag,
                                                       const _Float16* __restrict__ Bfrag,
                                                       const float* __restrict__ window,
                                                       float* __restrict__ frames) {
    __shared__ float lds[64 * 133];      // stride 133 -> bank-conflict-free transpose

    const int tid  = (int)threadIdx.x;
    const int lane = tid & 31;
    const int wv   = tid >> 5;           // wave id 0..7 (wave32)
    const int mg   = wv >> 1;            // 0..3 : which 32-row slice
    const int ng   = wv & 1;             // 0..1 : which 32-col slice

    const int mblock = (int)blockIdx.y;  // 0..7   (128 rows each)
    const int cblock = (int)blockIdx.x;  // 0..511 (64 cols each)

    const int mtile0 = mblock * 8 + mg * 2;   // global 16-row tile index
    const int ctile0 = cblock * 4 + ng * 2;   // global 16-col tile index

    v8f acc[2][2] = {};

    const _Float16* aB0 = Afrag + (size_t)(mtile0    ) * KTILES * FRAG_ELTS + lane * 16;
    const _Float16* aB1 = Afrag + (size_t)(mtile0 + 1) * KTILES * FRAG_ELTS + lane * 16;
    const _Float16* bB0 = Bfrag + (size_t)(ctile0    ) * KTILES * FRAG_ELTS + lane * 16;
    const _Float16* bB1 = Bfrag + (size_t)(ctile0 + 1) * KTILES * FRAG_ELTS + lane * 16;

    for (int kt = 0; kt < KTILES; ++kt) {
        const size_t off = (size_t)kt * FRAG_ELTS;
        v16h a0 = *(const v16h*)(aB0 + off);
        v16h a1 = *(const v16h*)(aB1 + off);
        v16h b0 = *(const v16h*)(bB0 + off);
        v16h b1 = *(const v16h*)(bB1 + off);

        // prefetch next K tile (gfx1250 global_prefetch_b8); speculative, OOB-safe
        const size_t noff = off + FRAG_ELTS;
        __builtin_prefetch((const void*)(aB0 + noff), 0, 1);
        __builtin_prefetch((const void*)(bB0 + noff), 0, 1);
        __builtin_prefetch((const void*)(bB1 + noff), 0, 1);

        acc[0][0] = __builtin_amdgcn_wmma_f32_16x16x32_f16(false, a0, false, b0, (short)0, acc[0][0], false, false);
        acc[0][1] = __builtin_amdgcn_wmma_f32_16x16x32_f16(false, a0, false, b1, (short)0, acc[0][1], false, false);
        acc[1][0] = __builtin_amdgcn_wmma_f32_16x16x32_f16(false, a1, false, b0, (short)0, acc[1][0], false, false);
        acc[1][1] = __builtin_amdgcn_wmma_f32_16x16x32_f16(false, a1, false, b1, (short)0, acc[1][1], false, false);
    }

    // D-tile layout (32-bit C/D 16x16): VGPR r, lanes0-15 -> M=r, N=lane ; lanes16-31 -> M=8+r, N=lane-16
    const int ml0 = mg * 32 + ((lane < 16) ? 0 : 8);
    const int cl0 = ng * 32 + (lane & 15);
    #pragma unroll
    for (int i = 0; i < 2; ++i)
        #pragma unroll
        for (int jj = 0; jj < 2; ++jj)
            #pragma unroll
            for (int r = 0; r < 8; ++r)
                lds[(cl0 + jj * 16) * 133 + (ml0 + i * 16 + r)] = acc[i][jj][r];
    __syncthreads();

    // cooperative coalesced store: 128x64 tile, contiguous 512B runs per column
    const int mbase = mblock * 128;
    const int cbase = cblock * 64;
    const float inv_nfft = 1.0f / 1024.0f;
    #pragma unroll 4
    for (int it = 0; it < 32; ++it) {
        const int g  = tid + it * 256;
        const int cl = g >> 7;           // 0..63
        const int m  = g & 127;          // 0..127
        const float wn = window[mbase + m];
        frames[(size_t)(cbase + cl) * NFFT + mbase + m] = lds[cl * 133 + m] * wn * inv_nfft;
    }
}

// ---------------- kernel 4: overlap-add + exact envelope normalization ----------------
__global__ __launch_bounds__(256) void istft_overlap_add(const float* __restrict__ frames,
                                                         const float* __restrict__ window,
                                                         float* __restrict__ out) {
    const size_t gid = (size_t)blockIdx.x * 256 + threadIdx.x;   // 16 * 524288 total
    const int b = (int)(gid >> 19);
    const int s = (int)(gid & 0x7FFFF);
    const int p = s + PAD;
    const int tmax = p >> 8;             // / HOP

    float acc = 0.0f, env = 0.0f;
    #pragma unroll
    for (int dj = 0; dj < 4; ++dj) {
        const int t = tmax - dj;
        if ((unsigned)t < (unsigned)TFRAMES) {
            const int n = p - (t << 8);  // always in [0, 1024)
            acc += frames[((size_t)b * TFRAMES + t) * NFFT + n];
            const float wn = window[n];
            env += wn * wn;
        }
    }
    out[gid] = acc / env;
}

// ---------------- launch ----------------
extern "C" void kernel_launch(void* const* d_in, const int* in_sizes, int n_in,
                              void* d_out, int out_size, void* d_ws, size_t ws_size,
                              hipStream_t stream) {
    const float* spec_real = (const float*)d_in[0];
    const float* spec_imag = (const float*)d_in[1];
    const float* window    = (const float*)d_in[2];
    float* out = (float*)d_out;

    // workspace layout (all 256B-aligned):
    //   frames : 1024 * 32768 f32            = 134,217,728 B
    //   Afrag  : 64 * 33 * 512 f16           =   2,162,688 B
    //   Bfrag  : 2048 * 33 * 512 f16         =  69,206,016 B
    char* ws = (char*)d_ws;
    float*    frames = (float*)ws;
    _Float16* Afrag  = (_Float16*)(ws + (size_t)134217728);
    _Float16* Bfrag  = (_Float16*)(ws + (size_t)134217728 + 2162688);

    // 1) basis fragments: 64*33*512 = 1,081,344 elements -> 4224 blocks
    istft_build_basis<<<4224, 256, 0, stream>>>(Afrag);

    // 2) spectrum fragments: 2048*33*512 = 34,603,008 elements -> 135168 blocks
    istft_convert_spec<<<135168, 256, 0, stream>>>(spec_real, spec_imag, Bfrag);

    // 3) WMMA GEMM: grid 512 col-blocks x 8 row-blocks
    dim3 ggrid(512, 8, 1);
    istft_gemm_wmma<<<ggrid, 256, 0, stream>>>(Afrag, Bfrag, window, frames);

    // 4) overlap-add: 16*524288 = 8,388,608 samples -> 32768 blocks
    istft_overlap_add<<<32768, 256, 0, stream>>>(frames, window, out);
}
